// KRLST_50714973831437
// MI455X (gfx1250) — matrix operand order repaired and checked by reading, not compiled, not run on hardware
//
#include <hip/hip_runtime.h>

// ---------------- problem constants ----------------
#define MD   4096
#define DD   64
#define YD4  4

#define FFc      0.999f
#define OMFc     0.001f
#define SQFc     0.99949987493742f   // sqrt(0.999)
#define REGc     0.01f
#define JITc     1e-10f
#define INV2LS2  0.0078125f          // 1/(2*8*8)

// ---------------- workspace layout (float offsets) ----------------
#define OFF_A2     0            // M   : row norms of dictionary
#define OFF_KDX    4096         // M   : K_dx
#define OFF_Q      8192         // M   : q_ = Q @ K_dx
#define OFF_H      12288        // M   : h_ = sig @ q_
#define OFF_MUA    16384        // YD*(M+1) : mu_a
#define OFF_CRIT   32772        // M+1 : criterion
#define OFF_QS     36869        // M   : Qs
#define OFF_SCAL   40968        // scalars (see indices below)
#define OFF_IPRUNE 41000        // int slot
#define OFF_D16    41024        // M*D halves (16B aligned: 41024*4 % 16 == 0)
// SCAL indices
#define S_PU    0
#define S_NPV   1
#define S_PV    2
#define S_IPU   3
#define S_IPV   4
#define S_LOW   5
#define S_PM    6   // +yd
#define S_R     10  // +yd
#define S_S     14  // +yd
#define S_DEN   18
#define S_IDEN  19

typedef _Float16 half_t;
typedef __attribute__((ext_vector_type(16))) _Float16 v16h;
typedef __attribute__((ext_vector_type(8)))  _Float16 v8h;
typedef __attribute__((ext_vector_type(8)))  float    v8f;

union V16H { v16h v; v8h h[2]; };

__device__ __forceinline__ float blockReduceSum(float v, float* red) {
    int tid = threadIdx.x;
    __syncthreads();
    red[tid] = v;
    __syncthreads();
    for (int s = 128; s > 0; s >>= 1) {
        if (tid < s) red[tid] += red[tid + s];
        __syncthreads();
    }
    return red[0];
}

// K1: dictionary row norms, f16 copy of dictionary, K_dx (exact f32 RBF to x)
__global__ void k_prep(const float* __restrict__ dict, const float* __restrict__ x,
                       const float* __restrict__ ti, float* ws, half_t* __restrict__ d16) {
    int lane = threadIdx.x & 31;
    int wave = threadIdx.x >> 5;
    int row  = blockIdx.x * 8 + wave;         // 512 blocks * 8 waves = 4096 rows
    float d0 = dict[row * DD + lane];
    float d1 = dict[row * DD + 32 + lane];
    d16[row * DD + lane]      = (half_t)d0;
    d16[row * DD + 32 + lane] = (half_t)d1;
    float x0 = x[lane], x1 = x[32 + lane];
    float a2 = d0 * d0 + d1 * d1;
    float xd = d0 * x0 + d1 * x1;
    float x2 = x0 * x0 + x1 * x1;
    for (int m = 1; m < 32; m <<= 1) {        // wave32 reduction
        a2 += __shfl_xor(a2, m, 32);
        xd += __shfl_xor(xd, m, 32);
        x2 += __shfl_xor(x2, m, 32);
    }
    if (lane == 0) {
        ws[OFF_A2 + row] = a2;
        float dd2 = fmaxf(a2 + x2 - 2.0f * xd, 0.0f);
        float kv  = __expf(-dd2 * INV2LS2);
        ws[OFF_KDX + row] = (ti[row] >= 0.0f) ? kv : 0.0f;
    }
}

// K2: q = Q @ K_dx  (one block per row, float4 streaming)
__global__ void k_gemv_q(const float* __restrict__ Q, float* ws) {
    __shared__ float red[256];
    int i = blockIdx.x;
    const float* qr  = Q + (size_t)i * MD;
    const float* kdx = ws + OFF_KDX;
    float acc = 0.f;
    for (int j = threadIdx.x * 4; j < MD; j += 1024) {
        __builtin_prefetch(qr + j + 2048, 0, 0);
        float4 v = *(const float4*)(qr + j);
        acc += v.x * kdx[j] + v.y * kdx[j + 1] + v.z * kdx[j + 2] + v.w * kdx[j + 3];
    }
    float s = blockReduceSum(acc, red);
    if (threadIdx.x == 0) ws[OFF_Q + i] = s;
}

// K3: sig = FF*sigma + (1-FF)*Kdd (WMMA f16 cross term + exact f32 norms),
//     fused h = sig @ q.  One 16-row strip per workgroup; sig written once
//     straight into d_out's sig_n slot (transformed in-place later).
//     Two independent WMMA accumulators (no D->C hazard NOPs) and the next
//     tile's B fragments are loaded before the current tile's math.
__global__ void k_sig_h(const float* __restrict__ sigma, const float* __restrict__ ti,
                        const half_t* __restrict__ d16, float* ws,
                        float* __restrict__ sig_out) {
    __shared__ float hred[16][8];
    int lane = threadIdx.x & 31;
    int wave = threadIdx.x >> 5;
    int hi   = lane >> 4;
    int nn   = lane & 15;
    int i0   = blockIdx.x * 16;

    // A fragment (16x32 f16, ISA layout): lane m=nn, K-chunks at kb and kb+16
    int arow = i0 + nn;
    int kb   = hi * 8;
    V16H alo, ahi;
    alo.h[0] = *(const v8h*)(d16 + arow * DD + kb);
    alo.h[1] = *(const v8h*)(d16 + arow * DD + 16 + kb);
    ahi.h[0] = *(const v8h*)(d16 + arow * DD + 32 + kb);
    ahi.h[1] = *(const v8h*)(d16 + arow * DD + 48 + kb);

    const float* A2 = ws + OFF_A2;
    const float* QV = ws + OFF_Q;
    float a2i[8], mi[8], hacc[8];
#pragma unroll
    for (int r = 0; r < 8; ++r) {
        int i  = i0 + r + 8 * hi;
        a2i[r] = A2[i];
        mi[r]  = (ti[i] >= 0.0f) ? 1.0f : 0.0f;
        hacc[r] = 0.0f;
    }

    const int ntiles = MD / 16;   // 256 column tiles
    int kb2 = hi * 8;   // B fragment (32x16): lanes 0-15 K=0..7,16..23, lanes 16-31 K=8..15,24..31

    // preload first B fragments for this wave
    V16H blo, bhi;
    {
        int brow = wave * 16 + nn;
        const v8h* pb  = (const v8h*)(d16 + brow * DD + kb2);
        const v8h* pb2 = (const v8h*)(d16 + brow * DD + 32 + kb2);
        blo.h[0] = pb[0];  blo.h[1] = pb[2];
        bhi.h[0] = pb2[0]; bhi.h[1] = pb2[2];
    }

    for (int jt = wave; jt < ntiles; jt += 8) {
        // branchless prefetch of next tile's B (index clamped; EXEC stays full)
        int jnext = (jt + 8 < ntiles) ? (jt + 8) : wave;
        V16H nblo, nbhi;
        {
            int brow = jnext * 16 + nn;
            const v8h* pb  = (const v8h*)(d16 + brow * DD + kb2);
            const v8h* pb2 = (const v8h*)(d16 + brow * DD + 32 + kb2);
            nblo.h[0] = pb[0];  nblo.h[1] = pb[2];
            nbhi.h[0] = pb2[0]; nbhi.h[1] = pb2[2];
        }

        // independent accumulators: both WMMAs take inline-0 C, no RAW hazard
        v8f c0 = {}, c1 = {};
        c0 = __builtin_amdgcn_wmma_f32_16x16x32_f16(false, alo.v, false, blo.v,
                                                    (short)0, c0, false, false);
        c1 = __builtin_amdgcn_wmma_f32_16x16x32_f16(false, ahi.v, false, bhi.v,
                                                    (short)0, c1, false, false);

        int j    = jt * 16 + nn;
        float a2j = A2[j];
        float mj  = (ti[j] >= 0.0f) ? 1.0f : 0.0f;
        float qj  = QV[j];
#pragma unroll
        for (int r = 0; r < 8; ++r) {
            size_t idx = (size_t)(i0 + r + 8 * hi) * MD + j;
            float dot = c0[r] + c1[r];
            float dd2 = fmaxf(a2i[r] + a2j - 2.0f * dot, 0.0f);
            float kdd = __expf(-dd2 * INV2LS2) * mi[r] * mj;
            float sv  = FFc * sigma[idx] + OMFc * kdd;
            sig_out[idx] = sv;
            hacc[r] += sv * qj;
        }
        blo = nblo;
        bhi = nbhi;
    }
    // reduce h over the 16 column-lanes of each half, then across 8 waves
#pragma unroll
    for (int r = 0; r < 8; ++r) {
        float v = hacc[r];
        v += __shfl_xor(v, 1, 32);
        v += __shfl_xor(v, 2, 32);
        v += __shfl_xor(v, 4, 32);
        v += __shfl_xor(v, 8, 32);
        if (nn == 0) hred[r + 8 * hi][wave] = v;
    }
    __syncthreads();
    if (threadIdx.x < 16) {
        float s = 0.f;
#pragma unroll
        for (int w = 0; w < 8; ++w) s += hred[threadIdx.x][w];
        ws[OFF_H + i0 + threadIdx.x] = s;
    }
}

// K4: all scalars (pu, npv, pv, pred_mean, r, s), crit[M], nnum, nden
__global__ void k_scalars(const float* __restrict__ mu, const float* __restrict__ y,
                          const float* __restrict__ nnum_in, const float* __restrict__ nden_in,
                          float* ws, float* __restrict__ nnum_out, float* __restrict__ nden_out) {
    __shared__ float red[256];
    const float* KDX = ws + OFF_KDX;
    const float* QV  = ws + OFF_Q;
    const float* H   = ws + OFF_H;
    float acc[6] = {0, 0, 0, 0, 0, 0};
    for (int j = threadIdx.x; j < MD; j += 256) {
        float q = QV[j];
        acc[0] += KDX[j] * q;
        acc[1] += H[j] * q;
        acc[2] += mu[0 * MD + j] * q;
        acc[3] += mu[1 * MD + j] * q;
        acc[4] += mu[2 * MD + j] * q;
        acc[5] += mu[3 * MD + j] * q;
    }
    float sums[6];
    for (int r = 0; r < 6; ++r) sums[r] = blockReduceSum(acc[r], red);
    if (threadIdx.x == 0) {
        float Kxx = 1.0f + JITc;                  // rbf(x,x) + jitter
        float pu  = fmaxf(Kxx - sums[0], 0.0f);
        float qh  = sums[1];
        float npv = fmaxf(pu + qh, 0.0f);
        float pv  = REGc + npv;
        ws[OFF_SCAL + S_PU]  = pu;
        ws[OFF_SCAL + S_NPV] = npv;
        ws[OFF_SCAL + S_PV]  = pv;
        ws[OFF_SCAL + S_IPU] = 1.0f / pu;
        ws[OFF_SCAL + S_IPV] = 1.0f / pv;
        ws[OFF_SCAL + S_LOW] = (pu < JITc) ? 1.0f : 0.0f;
        float critM = 0.f;
        for (int d = 0; d < 4; ++d) {
            float pm = SQFc * sums[2 + d];
            float rr = (y[d] - pm) / pv;
            float ss = rr * (qh - npv);           // = p2 . mu_a[d]
            ws[OFF_SCAL + S_PM + d] = pm;
            ws[OFF_SCAL + S_R + d]  = rr;
            ws[OFF_SCAL + S_S + d]  = ss;
            critM += fabsf(ss);                   // closed form of criterion[M]
            nnum_out[d] = nnum_in[d] + FFc * (y[d] - pm) * (y[d] - pm) / pv;
        }
        ws[OFF_CRIT + MD] = critM;
        nden_out[0] = nden_in[0] + FFc;
    }
}

// K5: mu_a = [sqrt(FF)*mu; pm] + r * [h; npv]
__global__ void k_mua(const float* __restrict__ mu, float* ws) {
    int idx = blockIdx.x * 256 + threadIdx.x;
    if (idx >= YD4 * (MD + 1)) return;
    int yd = idx / (MD + 1);
    int j  = idx - yd * (MD + 1);
    float rr = ws[OFF_SCAL + S_R + yd];
    float v;
    if (j < MD) v = SQFc * mu[yd * MD + j] + rr * ws[OFF_H + j];
    else        v = ws[OFF_SCAL + S_PM + yd] + rr * ws[OFF_SCAL + S_NPV];
    ws[OFF_MUA + idx] = v;
}

// K6: criterion rows 0..M-1: |Q_a @ mu_a| summed over yd, / |diag| — using
// Q_a = pad(Q) + p2 p2^T/pu so only Q itself is streamed (one block per row).
__global__ void k_crit(const float* __restrict__ Q, const float* __restrict__ ti, float* ws) {
    __shared__ float red[256];
    int i = blockIdx.x;
    const float* qr = Q + (size_t)i * MD;
    const float* m0 = ws + OFF_MUA;
    const float* m1 = m0 + (MD + 1);
    const float* m2 = m1 + (MD + 1);
    const float* m3 = m2 + (MD + 1);
    float a0 = 0, a1 = 0, a2 = 0, a3 = 0;
    for (int j = threadIdx.x * 4; j < MD; j += 1024) {
        __builtin_prefetch(qr + j + 2048, 0, 0);
        float4 v = *(const float4*)(qr + j);
        a0 += v.x * m0[j] + v.y * m0[j + 1] + v.z * m0[j + 2] + v.w * m0[j + 3];
        a1 += v.x * m1[j] + v.y * m1[j + 1] + v.z * m1[j + 2] + v.w * m1[j + 3];
        a2 += v.x * m2[j] + v.y * m2[j + 1] + v.z * m2[j + 2] + v.w * m2[j + 3];
        a3 += v.x * m3[j] + v.y * m3[j + 1] + v.z * m3[j + 2] + v.w * m3[j + 3];
    }
    float s0 = blockReduceSum(a0, red);
    float s1 = blockReduceSum(a1, red);
    float s2 = blockReduceSum(a2, red);
    float s3 = blockReduceSum(a3, red);
    if (threadIdx.x == 0) {
        float ipu = ws[OFF_SCAL + S_IPU];
        float qi  = ws[OFF_Q + i];
        float dq  = (ti[i] >= 0.0f) ? (qr[i] + qi * qi * ipu) : __builtin_inff();
        float num = fabsf(s0 + qi * ws[OFF_SCAL + S_S + 0] * ipu)
                  + fabsf(s1 + qi * ws[OFF_SCAL + S_S + 1] * ipu)
                  + fabsf(s2 + qi * ws[OFF_SCAL + S_S + 2] * ipu)
                  + fabsf(s3 + qi * ws[OFF_SCAL + S_S + 3] * ipu);
        ws[OFF_CRIT + i] = num * fabsf(1.0f / dq);
    }
}

// K7: argmin (first-index tiebreak, matching jnp.argmin) + denom
__global__ void k_argmin(const float* __restrict__ Q, const float* __restrict__ ti,
                         float* ws, int* __restrict__ iprune) {
    __shared__ float sv[256];
    __shared__ int   si[256];
    float bv = __builtin_inff(); int bi = MD + 1;
    for (int i = threadIdx.x; i <= MD; i += 256) {
        float v = ws[OFF_CRIT + i];
        if (v < bv || (v == bv && i < bi)) { bv = v; bi = i; }
    }
    sv[threadIdx.x] = bv; si[threadIdx.x] = bi;
    __syncthreads();
    for (int s = 128; s > 0; s >>= 1) {
        if (threadIdx.x < s) {
            float ov = sv[threadIdx.x + s]; int oi = si[threadIdx.x + s];
            if (ov < sv[threadIdx.x] || (ov == sv[threadIdx.x] && oi < si[threadIdx.x])) {
                sv[threadIdx.x] = ov; si[threadIdx.x] = oi;
            }
        }
        __syncthreads();
    }
    if (threadIdx.x == 0) {
        int p = si[0];
        if (ws[OFF_SCAL + S_LOW] > 0.5f) p = MD;   // crit_low branch -> argmin == M
        float ipu = ws[OFF_SCAL + S_IPU];
        float denom;
        if (p < MD) {
            float qp = ws[OFF_Q + p];
            denom = (ti[p] >= 0.0f) ? (Q[(size_t)p * MD + p] + qp * qp * ipu)
                                    : __builtin_inff();
        } else denom = ipu;                         // Q_a[M,M] = 1/pu
        iprune[0] = p;
        ws[OFF_SCAL + S_DEN]  = denom;
        ws[OFF_SCAL + S_IDEN] = 1.0f / denom;       // inf -> 0 (rank-1 vanishes)
    }
}

// K8: Qs[i] = Q_a[perm(i), prune]
__global__ void k_qs(const float* __restrict__ Q, float* ws, const int* __restrict__ iprune) {
    int i = blockIdx.x * 256 + threadIdx.x;
    int p = iprune[0];
    float ipu = ws[OFF_SCAL + S_IPU];
    const float* QV = ws + OFF_Q;
    float v;
    if (p < MD) v = (i == p) ? (-QV[p] * ipu)
                             : (Q[(size_t)i * MD + p] + QV[i] * QV[p] * ipu);
    else        v = -QV[i] * ipu;
    ws[OFF_QS + i] = v;
}

// K9: Q_n = perm(Q_a)[:M,:M] - Qs Qs^T / denom, elementwise from Q (single pass)
__global__ void k_qn(const float* __restrict__ Q, const float* ws,
                     const int* __restrict__ iprune, float* __restrict__ Qn) {
    size_t idx = (size_t)blockIdx.x * 256 + threadIdx.x;
    int i  = (int)(idx >> 10);           // / (M/4)
    int j0 = (int)(idx & 1023) * 4;
    int p = iprune[0];
    float ipu  = ws[OFF_SCAL + S_IPU];
    float iden = ws[OFF_SCAL + S_IDEN];
    const float* QV = ws + OFF_Q;
    const float* QS = ws + OFF_QS;
    float qi = QV[i], qsi = QS[i];
    float4 qv = *(const float4*)(Q + (size_t)i * MD + j0);
    float out[4];
#pragma unroll
    for (int e = 0; e < 4; ++e) {
        int j = j0 + e;
        float qe = (&qv.x)[e];
        bool ip = (i == p), jp = (j == p);
        float base;
        if (!ip && !jp)     base = qe + qi * QV[j] * ipu;   // Q_a[i,j]
        else if (ip && jp)  base = ipu;                     // Q_a[M,M]
        else if (ip)        base = -QV[j] * ipu;            // Q_a[M,j]
        else                base = -qi * ipu;               // Q_a[i,M]
        out[e] = base - qsi * QS[j] * iden;
    }
    *(float4*)(Qn + (size_t)i * MD + j0) = make_float4(out[0], out[1], out[2], out[3]);
}

// K10: sig_n in-place over the sig already stored in d_out:
// sig_n = mask * (perm(sig_a)[:M,:M]), sig_a = [[sig,h],[h^T,npv]] - p p^T/pv
__global__ void k_sign(const float* __restrict__ ti, const float* ws,
                       const int* __restrict__ iprune, const int* __restrict__ tptr,
                       float* sig) {
    size_t idx = (size_t)blockIdx.x * 256 + threadIdx.x;
    int i  = (int)(idx >> 10);
    int j0 = (int)(idx & 1023) * 4;
    int p = iprune[0];
    float tval = (float)tptr[0];
    float npv  = ws[OFF_SCAL + S_NPV];
    float ipv  = ws[OFF_SCAL + S_IPV];
    const float* H = ws + OFF_H;
    bool ip = (p < MD) && (i == p);
    float tii = ip ? tval : ti[i];
    float mi  = (tii >= 0.0f) ? 1.0f : 0.0f;
    float Pi  = ip ? npv : H[i];
    float Si_h = ip ? 0.0f : H[i];   // used when column is the pruned one
    float4 sv = *(const float4*)(sig + (size_t)i * MD + j0);
    float out[4];
#pragma unroll
    for (int e = 0; e < 4; ++e) {
        int j = j0 + e;
        bool jp = (p < MD) && (j == p);
        float tjj = jp ? tval : ti[j];
        float mj  = (tjj >= 0.0f) ? 1.0f : 0.0f;
        float Pj  = jp ? npv : H[j];
        float S;
        if (!ip && !jp)     S = (&sv.x)[e];   // sig[i,j]
        else if (ip && jp)  S = npv;          // sig_a[M,M]
        else if (ip)        S = H[j];         // sig_a[M,j] = h[j]
        else                S = Si_h;         // sig_a[i,M] = h[i]
        out[e] = mi * mj * (S - Pi * Pj * ipv);
    }
    *(float4*)(sig + (size_t)i * MD + j0) = make_float4(out[0], out[1], out[2], out[3]);
}

// K11: mu_n, dict_n, ti_n
__global__ void k_tail(const float* __restrict__ dict, const float* __restrict__ x,
                       const float* __restrict__ ti, const int* __restrict__ tptr,
                       const float* ws, const int* __restrict__ iprune,
                       float* __restrict__ mun, float* __restrict__ dictn,
                       float* __restrict__ tin) {
    int e = blockIdx.x * 256 + threadIdx.x;     // covers M*D = 262144
    int p = iprune[0];
    float tval = (float)tptr[0];
    if (e < MD * DD) {
        int i = e >> 6, d = e & 63;
        dictn[e] = (p < MD && i == p) ? x[d] : dict[e];
    }
    if (e < MD) tin[e] = (p < MD && e == p) ? tval : ti[e];
    if (e < YD4 * MD) {
        int yd = e >> 12, i = e & (MD - 1);
        bool ip = (p < MD && i == p);
        float tn = ip ? tval : ti[i];
        int  a  = ip ? MD : i;
        mun[e] = (tn < 0.0f) ? 0.0f : ws[OFF_MUA + yd * (MD + 1) + a];
    }
}

extern "C" void kernel_launch(void* const* d_in, const int* in_sizes, int n_in,
                              void* d_out, int out_size, void* d_ws, size_t ws_size,
                              hipStream_t stream) {
    const float* dict    = (const float*)d_in[0];
    const float* mu      = (const float*)d_in[1];
    const float* sigma   = (const float*)d_in[2];
    const float* Q       = (const float*)d_in[3];
    const float* x       = (const float*)d_in[4];
    const float* y       = (const float*)d_in[5];
    const float* nnum_in = (const float*)d_in[6];
    const float* nden_in = (const float*)d_in[7];
    const float* ti      = (const float*)d_in[8];
    const int*   tptr    = (const int*)d_in[9];

    float* out    = (float*)d_out;
    float* o_mu   = out;                                   // YD*M
    float* o_sig  = o_mu + (size_t)YD4 * MD;               // M*M
    float* o_Q    = o_sig + (size_t)MD * MD;               // M*M
    float* o_dict = o_Q + (size_t)MD * MD;                 // M*D
    float* o_ti   = o_dict + (size_t)MD * DD;              // M
    float* o_nnum = o_ti + MD;                             // YD
    float* o_nden = o_nnum + YD4;                          // 1

    float*  ws     = (float*)d_ws;
    int*    iprune = (int*)(ws + OFF_IPRUNE);
    half_t* d16    = (half_t*)(ws + OFF_D16);

    k_prep   <<<512,   256, 0, stream>>>(dict, x, ti, ws, d16);
    k_gemv_q <<<MD,    256, 0, stream>>>(Q, ws);
    k_sig_h  <<<MD/16, 256, 0, stream>>>(sigma, ti, d16, ws, o_sig);
    k_scalars<<<1,     256, 0, stream>>>(mu, y, nnum_in, nden_in, ws, o_nnum, o_nden);
    k_mua    <<<(YD4*(MD+1)+255)/256, 256, 0, stream>>>(mu, ws);
    k_crit   <<<MD,    256, 0, stream>>>(Q, ti, ws);
    k_argmin <<<1,     256, 0, stream>>>(Q, ti, ws, iprune);
    k_qs     <<<MD/256,256, 0, stream>>>(Q, ws, iprune);
    k_qn     <<<(MD/4)*MD/256, 256, 0, stream>>>(Q, ws, iprune, o_Q);
    k_sign   <<<(MD/4)*MD/256, 256, 0, stream>>>(ti, ws, iprune, tptr, o_sig);
    k_tail   <<<MD*DD/256, 256, 0, stream>>>(dict, x, ti, tptr, ws, iprune,
                                             o_mu, o_dict, o_ti);
}
